// PCNNAtt_23965917512370
// MI455X (gfx1250) — matrix operand order repaired and checked by reading, not compiled, not run on hardware
//
#include <hip/hip_runtime.h>
#include <hip/hip_bf16.h>

// ---------------------------------------------------------------------------
// PCNN+ATT fused forward for gfx1250 (MI455X), wave32 + WMMA bf16.
//
// Dims (fixed by reference):
//   N=4096 sentences, L=120, B=1024 bags (4 sent/bag), C=53 relations
//   NF=230 filters, K=3 taps, D_IN=60  ->  im2col GEMM 120x180 * 180x230
// Padded GEMM: M=128 (8 tiles), N=240 (15 tiles), K=192 (6 x 32)
//
// Round 3: weight panel staged in LDS (CDNA5 320KB LDS) so the WMMA inner
// loop feeds from ds_load_b128 instead of L2 round-trips.
// ---------------------------------------------------------------------------

typedef __attribute__((ext_vector_type(16))) __bf16 v16bf;
typedef __attribute__((ext_vector_type(8)))  float  v8f;

#define NSENT   4096
#define SEQL    120
#define NBAG    1024
#define CREL    53
#define NFILT   230
#define DIN     60
#define FEATD   690          // 3*NFILT
#define NPAD    240          // 15 * 16
#define KPAD    192          // 6 * 32
#define KSTRIDE 200          // LDS row stride for B panel (400B = 16*25: aligned + conflict-free)
#define XROWS   132          // 1 (left pad) + 120 + pad rows, covers m+tap <= 129
#define XSTRIDE 72           // LDS row stride in bf16 elems (bank-conflict-free)
#define OVERLAY_BYTES (NPAD * KSTRIDE * 2)   // 96000: max(sBt, sPart)

union FragAB { v16bf v; unsigned short u[16]; uint4 q[2]; };
union AccU   { v8f v; float f[8]; };

__device__ __forceinline__ unsigned short bf16_bits(float f) {
    unsigned u = __float_as_uint(f);
    unsigned r = u + 0x7fffu + ((u >> 16) & 1u);   // round-to-nearest-even
    return (unsigned short)(r >> 16);
}

// ---------------------------------------------------------------------------
// Kernel 1: build pre-transposed bf16 weight panel Bt[NPAD][KPAD],
//           q = tap*64 + ch  (zero padded). conv_w layout is [NF][DIN][K].
// ---------------------------------------------------------------------------
__global__ void prep_bt_kernel(const float* __restrict__ conv_w,
                               unsigned short* __restrict__ bt) {
    int idx = blockIdx.x * blockDim.x + threadIdx.x;
    if (idx >= NPAD * KPAD) return;
    int n  = idx / KPAD;
    int q  = idx % KPAD;
    int t  = q >> 6;          // tap 0..2
    int ch = q & 63;          // channel 0..63
    float v = 0.0f;
    if (n < NFILT && ch < DIN)
        v = conv_w[(n * DIN + ch) * 3 + t];
    bt[idx] = bf16_bits(v);
}

// ---------------------------------------------------------------------------
// Kernel 2: fused embedding gather -> bf16 WMMA conv GEMM -> piecewise
//           max-pool (+bias, ReLU) -> feat[sent][690].  One block / sentence.
//           256 threads = 8 waves; wave w owns M-tile w (rows 16w..16w+15).
//           B panel staged in LDS; sBt/sPart share one overlay region.
// ---------------------------------------------------------------------------
__global__ __launch_bounds__(256)
void pcnn_conv_pool_kernel(const int*   __restrict__ token_ids,
                           const int*   __restrict__ head_pos,
                           const int*   __restrict__ tail_pos,
                           const int*   __restrict__ mask,
                           const float* __restrict__ token_emb,
                           const float* __restrict__ pos1_emb,
                           const float* __restrict__ pos2_emb,
                           const float* __restrict__ conv_b,
                           const unsigned short* __restrict__ bt,
                           float* __restrict__ feat) {
    __shared__ unsigned short sX[XROWS * XSTRIDE];   // bf16 im2col source (19008 B)
    __shared__ int   sMask[128];
    __shared__ int   sTok[SEQL], sHp[SEQL], sTp[SEQL];
    // Overlay: phase 1 = bf16 B panel [NPAD][KSTRIDE]; phase 2 = per-wave
    // piece maxes float[8][3][NPAD] (23040 B). 96000 B total.
    __shared__ __align__(16) unsigned char sOverlay[OVERLAY_BYTES];
    unsigned short* sBt   = (unsigned short*)sOverlay;
    float*          sPart = (float*)sOverlay;        // [w][p][n] -> ((w*3+p)*NPAD+n)

    const int sent = blockIdx.x;
    const int tid  = threadIdx.x;

    // ---- stage ids / mask ----
    if (tid < SEQL) {
        sTok[tid]  = token_ids[sent * SEQL + tid];
        sHp[tid]   = head_pos [sent * SEQL + tid];
        sTp[tid]   = tail_pos [sent * SEQL + tid];
        sMask[tid] = mask     [sent * SEQL + tid];
    }

    // ---- stage weight panel into LDS (stride change 192 -> 200) ----
    // 240 rows * 24 uint4 per row = 5760 16B copies.
    for (int idx = tid; idx < NPAD * 24; idx += 256) {
        int n = idx / 24;
        int c = idx % 24;
        *(uint4*)&sBt[n * KSTRIDE + c * 8] = *(const uint4*)&bt[n * KPAD + c * 8];
    }
    __syncthreads();

    // ---- gather embeddings -> padded bf16 xpad (row r <-> seq pos r-1) ----
    for (int idx = tid; idx < XROWS * 64; idx += 256) {
        int r  = idx >> 6;
        int ch = idx & 63;
        float v = 0.0f;
        if (r >= 1 && r <= SEQL && ch < DIN) {
            int pos = r - 1;
            if (ch < 50)      v = token_emb[sTok[pos] * 50 + ch];
            else if (ch < 55) v = pos1_emb [sHp[pos]  * 5  + (ch - 50)];
            else              v = pos2_emb [sTp[pos]  * 5  + (ch - 55)];
        }
        sX[r * XSTRIDE + ch] = bf16_bits(v);
    }
    __syncthreads();

    // ---- WMMA GEMM: 8 waves x 15 N-tiles x 6 K-steps, all-LDS operands ----
    const int wave = tid >> 5;          // M-tile
    const int lane = tid & 31;
    const int lm   = lane & 15;         // M offset (A) / N offset (B,C)
    const int lh   = lane >> 4;         // K-half selector

    AccU acc[15];
#pragma unroll
    for (int i = 0; i < 15; ++i)
#pragma unroll
        for (int j = 0; j < 8; ++j) acc[i].f[j] = 0.0f;

    auto loadB = [&](FragAB& b, int s, int nt) {
        const int off = (nt * 16 + lm) * KSTRIDE + s * 32 + lh * 16;
        b.q[0] = *(const uint4*)&sBt[off];
        b.q[1] = *(const uint4*)&sBt[off + 8];
    };
    auto loadA = [&](FragAB& a, int s) {
        const int tap = s >> 1;
        const int chb = (s & 1) * 32;
        const int abase = (wave * 16 + lm + tap) * XSTRIDE + chb + lh * 16;
#pragma unroll
        for (int j = 0; j < 8; ++j) {
            a.u[j]     = sX[abase + j];
            a.u[8 + j] = sX[abase + 8 + j];
        }
    };

    FragAB a[2], b[3];
    loadA(a[0], 0);
    loadB(b[0], 0, 0);
    loadB(b[1], 0, 1);

#pragma unroll
    for (int it = 0; it < 90; ++it) {
        const int s  = it / 15;
        const int nt = it % 15;
        const int pf = it + 2;                     // B prefetch distance 2
        if (pf < 90) loadB(b[pf % 3], pf / 15, pf % 15);
        if (nt == 0 && s < 5) loadA(a[(s + 1) & 1], s + 1);  // next A early
        acc[nt].v = __builtin_amdgcn_wmma_f32_16x16x32_bf16(
            false, a[s & 1].v, false, b[it % 3].v, (short)0, acc[nt].v,
            false, false);
    }
    __syncthreads();   // all sBt reads done -> overlay region becomes sPart

    // ---- piecewise max over this wave's 16 rows (C layout: M=v+8*lh, N=lm) ----
#pragma unroll
    for (int nt = 0; nt < 15; ++nt) {
        float pm0 = -1e30f, pm1 = -1e30f, pm2 = -1e30f;
#pragma unroll
        for (int v = 0; v < 8; ++v) {
            int m = wave * 16 + lh * 8 + v;
            if (m < SEQL) {
                float x = acc[nt].f[v];
                int p = sMask[m];
                pm0 = (p == 1) ? fmaxf(pm0, x) : pm0;
                pm1 = (p == 2) ? fmaxf(pm1, x) : pm1;
                pm2 = (p == 3) ? fmaxf(pm2, x) : pm2;
            }
        }
        pm0 = fmaxf(pm0, __shfl_xor(pm0, 16, 32));
        pm1 = fmaxf(pm1, __shfl_xor(pm1, 16, 32));
        pm2 = fmaxf(pm2, __shfl_xor(pm2, 16, 32));
        if (lh == 0) {
            int col = nt * 16 + lm;
            sPart[(wave * 3 + 0) * NPAD + col] = pm0;
            sPart[(wave * 3 + 1) * NPAD + col] = pm1;
            sPart[(wave * 3 + 2) * NPAD + col] = pm2;
        }
    }
    __syncthreads();

    // ---- reduce across waves, add bias, ReLU, store feat ----
    for (int j = tid; j < FEATD; j += 256) {
        int p = j / NFILT;
        int n = j % NFILT;
        float v = -1e30f;
#pragma unroll
        for (int w = 0; w < 8; ++w) v = fmaxf(v, sPart[(w * 3 + p) * NPAD + n]);
        v += conv_b[n];
        feat[sent * FEATD + j] = fmaxf(v, 0.0f);
    }
}

// ---------------------------------------------------------------------------
// Kernel 3: att[n][c] = feat[n] . dense_w[c]   (4096 x 53, len-690 dots)
// ---------------------------------------------------------------------------
__global__ __launch_bounds__(64)
void att_kernel(const float* __restrict__ feat,
                const float* __restrict__ dense_w,
                float* __restrict__ att) {
    __shared__ float sf[FEATD];
    const int sent = blockIdx.x;
    for (int f = threadIdx.x; f < FEATD; f += 64)
        sf[f] = feat[sent * FEATD + f];
    __syncthreads();
    const int c = threadIdx.x;
    if (c < CREL) {
        const float* w = dense_w + c * FEATD;
        float s = 0.0f;
        for (int f = 0; f < FEATD; ++f) s = fmaf(sf[f], w[f], s);
        att[sent * CREL + c] = s;
    }
}

// ---------------------------------------------------------------------------
// Kernel 4: per-bag selective attention + final softmax diagonal.
//   logits[b,c,k] = sum_i w_i[c] * att[n_i,k] + dense_b[k]   (algebraic
//   collapse of rep @ dense_w).  One block per bag, thread c handles class c.
// ---------------------------------------------------------------------------
__global__ __launch_bounds__(64)
void bag_finalize_kernel(const float* __restrict__ att,
                         const float* __restrict__ dense_b,
                         float* __restrict__ out) {
    __shared__ float sa[4][CREL];
    const int bag = blockIdx.x;
    for (int idx = threadIdx.x; idx < 4 * CREL; idx += 64) {
        int i = idx / CREL, c = idx % CREL;
        sa[i][c] = att[(bag * 4 + i) * CREL + c];
    }
    __syncthreads();
    const int c = threadIdx.x;
    if (c < CREL) {
        // softmax over the 4 sentences for class c
        float a0 = sa[0][c], a1 = sa[1][c], a2 = sa[2][c], a3 = sa[3][c];
        float m = fmaxf(fmaxf(a0, a1), fmaxf(a2, a3));
        float e0 = __expf(a0 - m), e1 = __expf(a1 - m);
        float e2 = __expf(a2 - m), e3 = __expf(a3 - m);
        float inv = 1.0f / (e0 + e1 + e2 + e3);
        float w0 = e0 * inv, w1 = e1 * inv, w2 = e2 * inv, w3 = e3 * inv;

        // pass 1: logits max (+ keep diagonal logit)
        float lgmax = -1e30f, lgc = 0.0f;
        for (int k = 0; k < CREL; ++k) {
            float lg = dense_b[k] + w0 * sa[0][k] + w1 * sa[1][k]
                                  + w2 * sa[2][k] + w3 * sa[3][k];
            if (k == c) lgc = lg;
            lgmax = fmaxf(lgmax, lg);
        }
        // pass 2: softmax denominator
        float den = 0.0f;
        for (int k = 0; k < CREL; ++k) {
            float lg = dense_b[k] + w0 * sa[0][k] + w1 * sa[1][k]
                                  + w2 * sa[2][k] + w3 * sa[3][k];
            den += __expf(lg - lgmax);
        }
        out[bag * CREL + c] = __expf(lgc - lgmax) / den;
    }
}

// ---------------------------------------------------------------------------
extern "C" void kernel_launch(void* const* d_in, const int* in_sizes, int n_in,
                              void* d_out, int out_size, void* d_ws, size_t ws_size,
                              hipStream_t stream) {
    const int*   token_ids = (const int*)  d_in[0];
    const int*   head_pos  = (const int*)  d_in[1];
    const int*   tail_pos  = (const int*)  d_in[2];
    const int*   mask      = (const int*)  d_in[3];
    /* seg_ids (d_in[4]) is contiguous repeat(arange(B), 4) by construction */
    const float* token_emb = (const float*)d_in[5];
    const float* pos1_emb  = (const float*)d_in[6];
    const float* pos2_emb  = (const float*)d_in[7];
    const float* conv_w    = (const float*)d_in[8];
    const float* conv_b    = (const float*)d_in[9];
    const float* dense_w   = (const float*)d_in[10];
    const float* dense_b   = (const float*)d_in[11];
    float* out = (float*)d_out;

    // workspace layout
    unsigned short* bt   = (unsigned short*)d_ws;                  // 92160 B
    float*          feat = (float*)((char*)d_ws + 131072);         // 4096*690*4
    float*          att  = (float*)((char*)d_ws + 131072 + (size_t)NSENT * FEATD * 4);

    prep_bt_kernel<<<(NPAD * KPAD + 255) / 256, 256, 0, stream>>>(conv_w, bt);

    pcnn_conv_pool_kernel<<<NSENT, 256, 0, stream>>>(
        token_ids, head_pos, tail_pos, mask,
        token_emb, pos1_emb, pos2_emb, conv_b, bt, feat);

    att_kernel<<<NSENT, 64, 0, stream>>>(feat, dense_w, att);

    bag_finalize_kernel<<<NBAG, 64, 0, stream>>>(att, dense_b, out);
}
